// ColWiseGateSelfAttention_21088289423632
// MI455X (gfx1250) — matrix operand order repaired and checked by reading, not compiled, not run on hardware
//
#include <hip/hip_runtime.h>
#include <hip/hip_bf16.h>

// ---------------------------------------------------------------------------
// ColWiseGateSelfAttention for MI455X (gfx1250, wave32, WMMA bf16)
// ---------------------------------------------------------------------------

typedef __attribute__((ext_vector_type(16))) __bf16 v16bf;
typedef __attribute__((ext_vector_type(8)))  __bf16 bf16x8;
typedef __attribute__((ext_vector_type(8)))  float  v8f;

static constexpr int Dd   = 1152;     // channels
static constexpr int D3d  = 384;      // channels per group
static constexpr int NTOK = 4 * 4096; // B*L = 16384 tokens

#define WMMA_BF16(a, b, c) \
  __builtin_amdgcn_wmma_f32_16x16x32_bf16(false, (a), false, (b), (short)0, (c), false, false)

// --- fp32 -> bf16 bulk convert (weights) -----------------------------------
__global__ __launch_bounds__(256) void k_f32_to_bf16(const float* __restrict__ s,
                                                     __bf16* __restrict__ d, int n) {
  for (int i = blockIdx.x * blockDim.x + threadIdx.x; i < n; i += gridDim.x * blockDim.x)
    d[i] = (__bf16)s[i];
}

// --- LayerNorm over D=1152, write bf16 row ---------------------------------
__global__ __launch_bounds__(256) void k_layernorm(const float* __restrict__ x,
                                                   const float* __restrict__ gamma,
                                                   const float* __restrict__ beta,
                                                   __bf16* __restrict__ xn) {
  const int row = blockIdx.x;                 // token
  const float* xr = x + (size_t)row * Dd;
  float s = 0.f, s2 = 0.f;
  for (int i = threadIdx.x; i < Dd; i += 256) { float v = xr[i]; s += v; s2 += v * v; }
  for (int off = 16; off > 0; off >>= 1) {
    s  += __shfl_down(s,  off, 32);
    s2 += __shfl_down(s2, off, 32);
  }
  __shared__ float red[2][8];
  const int wave = threadIdx.x >> 5, lane = threadIdx.x & 31;
  if (lane == 0) { red[0][wave] = s; red[1][wave] = s2; }
  __syncthreads();
  if (threadIdx.x == 0) {
    float ts = 0.f, ts2 = 0.f;
    for (int w = 0; w < 8; ++w) { ts += red[0][w]; ts2 += red[1][w]; }
    float mu  = ts / (float)Dd;
    float var = ts2 / (float)Dd - mu * mu;
    red[0][0] = mu;
    red[1][0] = rsqrtf(var + 1e-6f);
  }
  __syncthreads();
  const float mu = red[0][0], rstd = red[1][0];
  for (int i = threadIdx.x; i < Dd; i += 256)
    xn[(size_t)row * Dd + i] = (__bf16)((xr[i] - mu) * rstd * gamma[i] + beta[i]);
}

// --- fragment helpers -------------------------------------------------------
// A (16-bit, 16x32): lane l holds row (l&15); elems 0..7 = K[k0+8*kc .. +7],
// elems 8..15 = K[k0+16+8*kc .. +7], kc = l>>4.
__device__ __forceinline__ v16bf load_a_frag(const __bf16* arow, int k0, int kc) {
  bf16x8 lo = *(const bf16x8*)(arow + k0 + 8 * kc);
  bf16x8 hi = *(const bf16x8*)(arow + k0 + 16 + 8 * kc);
  v16bf a;
#pragma unroll
  for (int i = 0; i < 8; ++i) { a[i] = lo[i]; a[i + 8] = hi[i]; }
  return a;
}
// B (16-bit, 32x16): lane l holds column (l&15); elems 0..7 = K[k0+8*kc .. +7],
// elems 8..15 = K[k0+16+8*kc .. +7], kc = l>>4.
__device__ __forceinline__ v16bf load_b_frag(const __bf16* brow, int k0, int kc) {
  bf16x8 lo = *(const bf16x8*)(brow + k0 + 8 * kc);
  bf16x8 hi = *(const bf16x8*)(brow + k0 + 16 + 8 * kc);
  v16bf b;
#pragma unroll
  for (int i = 0; i < 8; ++i) { b[i] = lo[i]; b[i + 8] = hi[i]; }
  return b;
}

// --- fused Q/K/V/G projection + 3x3 gated attention -------------------------
// One workgroup: 16 tokens -> 48 (token,group) rows, 12 waves (384 threads).
__global__ __launch_bounds__(384) void k_proj_attn(
    const __bf16* __restrict__ xn,
    const __bf16* __restrict__ Wq, const __bf16* __restrict__ Wk,
    const __bf16* __restrict__ Wv, const __bf16* __restrict__ Wg,
    const float* __restrict__ bq, const float* __restrict__ bk,
    const float* __restrict__ bv, const float* __restrict__ bg,
    __bf16* __restrict__ hout) {
  constexpr int LDA = 392;                    // padded bf16 stride (784 B, 16B aligned)
  extern __shared__ __bf16 smem[];
  __bf16* lA = smem;                          // 48 x LDA  activations
  __bf16* lQ = lA + 48 * LDA;                 // 48 x LDA  Q (later: attention output O)
  __bf16* lK = lQ + 48 * LDA;                 // 48 x LDA  K
  __bf16* lV = lK + 48 * LDA;                 // 48 x LDA  V

  const int t0   = blockIdx.x * 16;           // first token of this block
  const int tid  = threadIdx.x;
  const int lane = tid & 31;
  const int wave = __builtin_amdgcn_readfirstlane(tid >> 5);  // 0..11, scalar
  const int fm   = lane & 15;
  const int kc   = lane >> 4;

  // stage A: 48 rows x 384 bf16, contiguous in xn (token rows = 3x384 groups)
  {
    const __bf16* src = xn + (size_t)t0 * Dd;
    for (int c = tid; c < 48 * 48; c += 384) {        // 16-byte chunks
      int row = c / 48, col8 = (c % 48) * 8;
      *(bf16x8*)(lA + row * LDA + col8) = *(const bf16x8*)(src + row * D3d + col8);
    }
  }
  __syncthreads();

  // Q,K,V GEMMs. p unrolled at compile time -> direct (scalar-base) weight
  // pointers; one B fragment feeds all 3 M-tiles (48 rows) per K-step.
#pragma unroll
  for (int p = 0; p < 3; ++p) {
    const __bf16* W  = (p == 0) ? Wq : (p == 1) ? Wk : Wv;
    const float*  bb = (p == 0) ? bq : (p == 1) ? bk : bv;
    __bf16*      dst = (p == 0) ? lQ : (p == 1) ? lK : lV;
    for (int nt = wave; nt < 24; nt += 12) {          // 2 n-tiles per wave
      const __bf16* wrow  = W + (size_t)(nt * 16 + fm) * D3d;
      const __bf16* arow0 = lA + fm * LDA;
      v8f a0 = {}, a1 = {}, a2 = {};
#pragma unroll 4
      for (int kk = 0; kk < 12; ++kk) {
        const int k0 = 32 * kk;
        const v16bf b = load_b_frag(wrow, k0, kc);
        a0 = WMMA_BF16(load_a_frag(arow0,            k0, kc), b, a0);
        a1 = WMMA_BF16(load_a_frag(arow0 + 16 * LDA, k0, kc), b, a1);
        a2 = WMMA_BF16(load_a_frag(arow0 + 32 * LDA, k0, kc), b, a2);
      }
      const int   col  = nt * 16 + fm;
      const float bias = bb[col];
#pragma unroll
      for (int r = 0; r < 8; ++r) {                   // C: M = r + 8*kc, N = fm
        const int row = r + 8 * kc;
        dst[(row     ) * LDA + col] = (__bf16)(a0[r] + bias);
        dst[(row + 16) * LDA + col] = (__bf16)(a1[r] + bias);
        dst[(row + 32) * LDA + col] = (__bf16)(a2[r] + bias);
      }
    }
  }
  __syncthreads();

  // 3x3 attention per (token, head): 256 active threads
  const int th_t = tid >> 4, th_h = tid & 15;
  float attn[3][3];
  if (tid < 256) {
    const float scale = 0.11785113019775793f;   // 1/sqrt(DK=72)
    float s[3][3];
#pragma unroll
    for (int i = 0; i < 3; ++i)
#pragma unroll
      for (int j = 0; j < 3; ++j) {
        const __bf16* qr = lQ + (3 * th_t + i) * LDA + th_h * 24;
        const __bf16* kr = lK + (3 * th_t + j) * LDA + th_h * 24;
        float a = 0.f;
#pragma unroll
        for (int c = 0; c < 24; ++c) a += (float)qr[c] * (float)kr[c];
        s[i][j] = a * scale;
      }
#pragma unroll
    for (int i = 0; i < 3; ++i) {
      const float mx = fmaxf(s[i][0], fmaxf(s[i][1], s[i][2]));
      const float e0 = __expf(s[i][0] - mx), e1 = __expf(s[i][1] - mx), e2 = __expf(s[i][2] - mx);
      const float inv = 1.f / (e0 + e1 + e2);
      attn[i][0] = e0 * inv; attn[i][1] = e1 * inv; attn[i][2] = e2 * inv;
    }
  }
  __syncthreads();                              // done reading Q/K
  if (tid < 256) {                              // O = attn @ V overwrites Q region
#pragma unroll
    for (int i = 0; i < 3; ++i)
#pragma unroll
      for (int c = 0; c < 24; ++c) {
        float o = 0.f;
#pragma unroll
        for (int j = 0; j < 3; ++j)
          o += attn[i][j] * (float)lV[(3 * th_t + j) * LDA + th_h * 24 + c];
        lQ[(3 * th_t + i) * LDA + th_h * 24 + c] = (__bf16)o;
      }
  }
  __syncthreads();

  // Gate GEMM fused with sigmoid gating + transposed store to h_out
  for (int nt = wave; nt < 24; nt += 12) {
    const __bf16* wrow  = Wg + (size_t)(nt * 16 + fm) * D3d;
    const __bf16* arow0 = lA + fm * LDA;
    v8f a0 = {}, a1 = {}, a2 = {};
#pragma unroll 4
    for (int kk = 0; kk < 12; ++kk) {
      const int k0 = 32 * kk;
      const v16bf b = load_b_frag(wrow, k0, kc);
      a0 = WMMA_BF16(load_a_frag(arow0,            k0, kc), b, a0);
      a1 = WMMA_BF16(load_a_frag(arow0 + 16 * LDA, k0, kc), b, a1);
      a2 = WMMA_BF16(load_a_frag(arow0 + 32 * LDA, k0, kc), b, a2);
    }
    const int   col  = nt * 16 + fm;            // col = h*24 + c
    const float bias = bg[col];
#pragma unroll
    for (int mt = 0; mt < 3; ++mt) {
      const v8f acc = (mt == 0) ? a0 : (mt == 1) ? a1 : a2;
#pragma unroll
      for (int r = 0; r < 8; ++r) {
        const int row = mt * 16 + r + 8 * kc;   // row = 3*t_local + g
        const float sg = 1.f / (1.f + __expf(-(acc[r] + bias)));
        const float ov = (float)lQ[row * LDA + col];
        const int token = t0 + row / 3, grp = row % 3;
        hout[(size_t)token * Dd + grp * D3d + col] = (__bf16)(ov * sg);
      }
    }
  }
}

// --- output GEMM: out = h @ Wo^T + bo + x * g  (M=16384, N=K=1152) ----------
// Workgroup: 32-row M-tile x 128-col N-tile, 8 waves; each wave: 2 M-tiles.
__global__ __launch_bounds__(256) void k_out_gemm(
    const __bf16* __restrict__ hA, const __bf16* __restrict__ Wo,
    const float* __restrict__ bo, const float* __restrict__ x,
    const float* __restrict__ g, float* __restrict__ out) {
  constexpr int LDA = 1160;                     // padded bf16 stride
  extern __shared__ __bf16 sA[];                // 32 x LDA
  const int m0 = blockIdx.x * 32;
  const int tid = threadIdx.x, lane = tid & 31;
  const int wave = __builtin_amdgcn_readfirstlane(tid >> 5);

  const __bf16* src = hA + (size_t)m0 * Dd;     // 32*1152 contiguous
  for (int c = tid; c < 32 * 144; c += 256) {   // 16-byte chunks
    int row = c / 144, col8 = (c % 144) * 8;
    *(bf16x8*)(sA + row * LDA + col8) = *(const bf16x8*)(src + row * Dd + col8);
  }
  __syncthreads();

  const int fm = lane & 15, kc = lane >> 4;
  const int n0 = blockIdx.y * 128 + wave * 16;
  const __bf16* wrow  = Wo + (size_t)(n0 + fm) * Dd;
  const __bf16* arow0 = sA + fm * LDA;
  v8f a0 = {}, a1 = {};
#pragma unroll 2
  for (int kk = 0; kk < 36; ++kk) {
    const int k0 = 32 * kk;
    __builtin_prefetch(wrow + k0 + 64, 0, 1);   // global_prefetch_b8 on weight stream
    const v16bf b = load_b_frag(wrow, k0, kc);
    a0 = WMMA_BF16(load_a_frag(arow0,            k0, kc), b, a0);
    a1 = WMMA_BF16(load_a_frag(arow0 + 16 * LDA, k0, kc), b, a1);
  }
  const float g0 = g[0];
  const int n = n0 + fm;
  const float bn = bo[n];
#pragma unroll
  for (int r = 0; r < 8; ++r) {
    const int m = m0 + r + 8 * kc;
    const size_t off0 = (size_t)m * Dd + n;
    const size_t off1 = (size_t)(m + 16) * Dd + n;
    out[off0] = a0[r] + bn + x[off0] * g0;
    out[off1] = a1[r] + bn + x[off1] * g0;
  }
}

// ---------------------------------------------------------------------------
extern "C" void kernel_launch(void* const* d_in, const int* in_sizes, int n_in,
                              void* d_out, int out_size, void* d_ws, size_t ws_size,
                              hipStream_t stream) {
  (void)in_sizes; (void)n_in; (void)out_size; (void)ws_size;
  const float* x  = (const float*)d_in[0];
  const float* lg = (const float*)d_in[1];
  const float* lb = (const float*)d_in[2];
  const float* Wq = (const float*)d_in[3];
  const float* bq = (const float*)d_in[4];
  const float* Wk = (const float*)d_in[5];
  const float* bk = (const float*)d_in[6];
  const float* Wv = (const float*)d_in[7];
  const float* bv = (const float*)d_in[8];
  const float* Wg = (const float*)d_in[9];
  const float* bg = (const float*)d_in[10];
  const float* Wo = (const float*)d_in[11];
  const float* bo = (const float*)d_in[12];
  const float* g  = (const float*)d_in[13];
  float* out = (float*)d_out;

  char* ws = (char*)d_ws;
  size_t off = 0;
  auto alloc = [&](size_t bytes) -> char* {
    char* p = ws + off; off += (bytes + 255) & ~(size_t)255; return p;
  };
  __bf16* xn   = (__bf16*)alloc((size_t)NTOK * Dd * sizeof(__bf16));
  __bf16* hbuf = (__bf16*)alloc((size_t)NTOK * Dd * sizeof(__bf16));
  __bf16* Wq_b = (__bf16*)alloc((size_t)D3d * D3d * sizeof(__bf16));
  __bf16* Wk_b = (__bf16*)alloc((size_t)D3d * D3d * sizeof(__bf16));
  __bf16* Wv_b = (__bf16*)alloc((size_t)D3d * D3d * sizeof(__bf16));
  __bf16* Wg_b = (__bf16*)alloc((size_t)D3d * D3d * sizeof(__bf16));
  __bf16* Wo_b = (__bf16*)alloc((size_t)Dd * Dd * sizeof(__bf16));

  k_f32_to_bf16<<<576,  256, 0, stream>>>(Wq, Wq_b, D3d * D3d);
  k_f32_to_bf16<<<576,  256, 0, stream>>>(Wk, Wk_b, D3d * D3d);
  k_f32_to_bf16<<<576,  256, 0, stream>>>(Wv, Wv_b, D3d * D3d);
  k_f32_to_bf16<<<576,  256, 0, stream>>>(Wg, Wg_b, D3d * D3d);
  k_f32_to_bf16<<<5184, 256, 0, stream>>>(Wo, Wo_b, Dd * Dd);

  k_layernorm<<<NTOK, 256, 0, stream>>>(x, lg, lb, xn);

  const size_t smem_attn = (size_t)4 * 48 * 392 * sizeof(__bf16); // ~150.5 KB
  k_proj_attn<<<NTOK / 16, 384, smem_attn, stream>>>(
      xn, Wq_b, Wk_b, Wv_b, Wg_b, bq, bk, bv, bg, hbuf);

  const size_t smem_out = (size_t)32 * 1160 * sizeof(__bf16);     // ~72.5 KB
  k_out_gemm<<<dim3(NTOK / 32, 9), 256, smem_out, stream>>>(
      hbuf, Wo_b, bo, x, g, out);
}
